// RBF_48421461295372
// MI455X (gfx1250) — matrix operand order repaired and checked by reading, not compiled, not run on hardware
//
#include <hip/hip_runtime.h>
#include <hip/hip_bf16.h>

typedef _Float16 h16;
typedef __attribute__((ext_vector_type(16))) _Float16 v16h;
typedef __attribute__((ext_vector_type(8)))  _Float16 v8h;
typedef __attribute__((ext_vector_type(8)))  float    v8f;

#define D_MODEL 512
#define N_PIX   1024
#define HEADS   8
#define DK      64
#define BS      8

#define KT_STRIDE 72   // halfs per K-tile row (144B, 16B-aligned, conflict-free)
#define VT_STRIDE 40   // halfs per V-tile row
#define KT_SZ (32 * KT_STRIDE)
#define VT_SZ (64 * VT_STRIDE)

static __device__ __forceinline__ v8h ld8(const h16* p) {
    return *(const v8h*)p;
}
static __device__ __forceinline__ v16h cat16(v8h a, v8h b) {
    return __builtin_shufflevector(a, b, 0,1,2,3,4,5,6,7,8,9,10,11,12,13,14,15);
}
static __device__ __forceinline__ v8f wmma_f16(v16h a, v16h b, v8f c) {
    return __builtin_amdgcn_wmma_f32_16x16x32_f16(false, a, false, b, (short)0, c, false, false);
}

// ---------------------------------------------------------------------------
// Projection GEMM: Y[b][o][p] = sum_c W[o][c] * X[b][c][p] + bias[o]
// MODE 0: X f32, out f16 transposed  [b][p][c]   (Q, K)
// MODE 1: X f32, out f16 normal      [b][c][p]   (V)
// MODE 2: X f16, out f32 normal      [b][c][p]   (final projection -> d_out)
// Block: 128 threads (4 waves), 64x64 output tile, K-step 32.
// ---------------------------------------------------------------------------
template <int MODE>
__global__ __launch_bounds__(128)
void proj_kernel(const void* __restrict__ Xin, const float* __restrict__ W,
                 const float* __restrict__ bias, void* __restrict__ Out) {
    __shared__ h16 Wt[64 * 40];   // [m][kk], 40-half stride (80B rows, 16B chunks)
    __shared__ h16 Xt[64 * 40];   // [n][kk] = X^T tile

    const int b   = blockIdx.z;
    const int m0  = blockIdx.y * 64;
    const int n0  = blockIdx.x * 64;
    const int t    = threadIdx.x;
    const int w    = t >> 5;
    const int lane = t & 31;
    const int row  = lane & 15;
    const int sel  = lane >> 4;

    v8f acc[4] = {};

    for (int k0 = 0; k0 < D_MODEL; k0 += 32) {
#pragma unroll
        for (int i = 0; i < 16; ++i) {
            int idx = t + 128 * i;
            int kk = idx & 31, m = idx >> 5;
            Wt[m * 40 + kk] = (h16)W[(size_t)(m0 + m) * D_MODEL + k0 + kk];
        }
#pragma unroll
        for (int i = 0; i < 16; ++i) {
            int idx = t + 128 * i;
            int n = idx & 63, kk = idx >> 6;
            float xv;
            if (MODE == 2)
                xv = (float)((const h16*)Xin)[((size_t)b * D_MODEL + k0 + kk) * N_PIX + n0 + n];
            else
                xv = ((const float*)Xin)[((size_t)b * D_MODEL + k0 + kk) * N_PIX + n0 + n];
            Xt[n * 40 + kk] = (h16)xv;
        }
        __syncthreads();

        const h16* ar = &Wt[(w * 16 + row) * 40 + sel * 8];
        v16h A = cat16(ld8(ar), ld8(ar + 16));
#pragma unroll
        for (int nt = 0; nt < 4; ++nt) {
            const h16* br = &Xt[(nt * 16 + row) * 40 + sel * 8];
            v16h B = cat16(ld8(br), ld8(br + 16));
            acc[nt] = wmma_f16(A, B, acc[nt]);
        }
        __syncthreads();
    }

    const int c0 = m0 + w * 16 + sel * 8;
    float bb[8];
#pragma unroll
    for (int i = 0; i < 8; ++i) bb[i] = bias[c0 + i];

#pragma unroll
    for (int nt = 0; nt < 4; ++nt) {
        const int p = n0 + nt * 16 + row;
        if (MODE == 0) {
            v8h pk;
#pragma unroll
            for (int i = 0; i < 8; ++i) pk[i] = (h16)(acc[nt][i] + bb[i]);
            *(v8h*)((h16*)Out + ((size_t)b * N_PIX + p) * D_MODEL + c0) = pk;
        } else if (MODE == 1) {
#pragma unroll
            for (int i = 0; i < 8; ++i)
                ((h16*)Out)[((size_t)b * D_MODEL + c0 + i) * N_PIX + p] = (h16)(acc[nt][i] + bb[i]);
        } else {
#pragma unroll
            for (int i = 0; i < 8; ++i)
                ((float*)Out)[((size_t)b * D_MODEL + c0 + i) * N_PIX + p] = acc[nt][i] + bb[i];
        }
    }
}

// Issue the 4 per-wave async copies staging one 32-key K tile and V tile.
static __device__ __forceinline__ void stage_kv(int t, int m0, int hc,
                                                unsigned long long kbase,
                                                unsigned long long vbase,
                                                h16* Kbuf, h16* Vbuf) {
#pragma unroll
    for (int j = 0; j < 2; ++j) {
        int id  = t + 128 * j;
        int key = id >> 3, c = id & 7;
        unsigned goff = (unsigned)((((m0 + key) * D_MODEL) + hc + c * 8) * 2);
        unsigned ldst = (unsigned)(size_t)&Kbuf[key * KT_STRIDE + c * 8];
        asm volatile("global_load_async_to_lds_b128 %0, %1, %2"
                     :: "v"(ldst), "v"(goff), "s"(kbase) : "memory");
    }
#pragma unroll
    for (int j = 0; j < 2; ++j) {
        int id = t + 128 * j;
        int d = id >> 2, c = id & 3;
        unsigned goff = (unsigned)((((hc + d) * N_PIX) + m0 + c * 8) * 2);
        unsigned ldst = (unsigned)(size_t)&Vbuf[d * VT_STRIDE + c * 8];
        asm volatile("global_load_async_to_lds_b128 %0, %1, %2"
                     :: "v"(ldst), "v"(goff), "s"(vbase) : "memory");
    }
}

// ---------------------------------------------------------------------------
// Flash attention per (b, h): streams 1024 keys in 32-key blocks.
// Double-buffered async K/V staging: tile i+1's 4 async copies are issued
// before computing tile i; s_wait_asynccnt 4 retires the older tile (in-order
// completion), fully overlapping HBM->LDS latency with WMMA + softmax.
// ---------------------------------------------------------------------------
__global__ __launch_bounds__(128)
void attn_kernel(const h16* __restrict__ Qt, const h16* __restrict__ Kt,
                 const h16* __restrict__ V, h16* __restrict__ Aw) {
    __shared__ h16 Ksh[2 * KT_SZ];      // double-buffered [key][d]
    __shared__ h16 Vsh[2 * VT_SZ];      // double-buffered [d][key]
    __shared__ h16 Plds[4 * 16 * 40];   // per-wave 16x32 prob tile

    const int b  = blockIdx.z;
    const int h  = blockIdx.y;
    const int qb = blockIdx.x;
    const int t    = threadIdx.x;
    const int w    = t >> 5;
    const int lane = t & 31;
    const int row  = lane & 15;
    const int sel  = lane >> 4;

    const int qBase = qb * 64 + w * 16;
    const int hc    = h * DK;
    h16* Pw = &Plds[w * 16 * 40];

    // Q operand (16 q-rows x 64 d), two K=32 chunks
    const h16* qr = Qt + ((size_t)b * N_PIX + qBase + row) * D_MODEL + hc;
    v16h aq0 = cat16(ld8(qr + sel * 8),      ld8(qr + 16 + sel * 8));
    v16h aq1 = cat16(ld8(qr + 32 + sel * 8), ld8(qr + 48 + sel * 8));

    v16h ones;
#pragma unroll
    for (int i = 0; i < 16; ++i) ones[i] = (h16)1.0f;

    float mst[8], lst[8];
#pragma unroll
    for (int i = 0; i < 8; ++i) { mst[i] = -1e30f; lst[i] = 0.f; }
    v8f acc[4] = {};
    const v8f zero = {};

    const unsigned long long kbase =
        (unsigned long long)(size_t)(Kt + (size_t)b * N_PIX * D_MODEL);
    const unsigned long long vbase =
        (unsigned long long)(size_t)(V + (size_t)b * D_MODEL * N_PIX);

    const int nTiles = N_PIX / 32;
    stage_kv(t, 0, hc, kbase, vbase, Ksh, Vsh);   // prologue: tile 0 -> buf 0

    for (int it = 0; it < nTiles; ++it) {
        const int m0 = it * 32;
        h16* Kc = Ksh + (it & 1) * KT_SZ;
        h16* Vc = Vsh + (it & 1) * VT_SZ;

        if (it + 1 < nTiles) {
            stage_kv(t, m0 + 32, hc, kbase, vbase,
                     Ksh + ((it + 1) & 1) * KT_SZ, Vsh + ((it + 1) & 1) * VT_SZ);
            asm volatile("s_wait_asynccnt 4" ::: "memory");  // tile `it` done
        } else {
            asm volatile("s_wait_asynccnt 0" ::: "memory");
        }
        __syncthreads();

        // --- scores: B operands from shared K tile
        const h16* k0p = &Kc[row * KT_STRIDE];
        const h16* k1p = &Kc[(row + 16) * KT_STRIDE];
        v16h bk00 = cat16(ld8(k0p + sel * 8),      ld8(k0p + sel * 8 + 16));
        v16h bk01 = cat16(ld8(k0p + 32 + sel * 8), ld8(k0p + 32 + sel * 8 + 16));
        v16h bk10 = cat16(ld8(k1p + sel * 8),      ld8(k1p + sel * 8 + 16));
        v16h bk11 = cat16(ld8(k1p + 32 + sel * 8), ld8(k1p + 32 + sel * 8 + 16));

        v8f s0 = wmma_f16(aq1, bk01, zero); s0 = wmma_f16(aq0, bk00, s0);
        v8f s1 = wmma_f16(aq1, bk11, zero); s1 = wmma_f16(aq0, bk10, s1);

        // scale 1/sqrt(64) and masked_fill(score==0, -1e9)
#pragma unroll
        for (int i = 0; i < 8; ++i) {
            s0[i] *= 0.125f; if (s0[i] == 0.f) s0[i] = -1e9f;
            s1[i] *= 0.125f; if (s1[i] == 0.f) s1[i] = -1e9f;
        }

        // per-row max over 32 new cols (xor-shuffle within each 16-lane half)
        float tm[8];
#pragma unroll
        for (int i = 0; i < 8; ++i) tm[i] = fmaxf(s0[i], s1[i]);
#pragma unroll
        for (int mk = 1; mk < 16; mk <<= 1)
#pragma unroll
            for (int i = 0; i < 8; ++i) tm[i] = fmaxf(tm[i], __shfl_xor(tm[i], mk, 32));

        float fac[8];
#pragma unroll
        for (int i = 0; i < 8; ++i) {
            float mn = fmaxf(mst[i], tm[i]);
            fac[i] = __expf(mst[i] - mn);
            mst[i] = mn;
        }
#pragma unroll
        for (int i = 0; i < 8; ++i) {
            s0[i] = __expf(s0[i] - mst[i]);
            s1[i] = __expf(s1[i] - mst[i]);
        }
#pragma unroll
        for (int nt = 0; nt < 4; ++nt)
#pragma unroll
            for (int i = 0; i < 8; ++i) acc[nt][i] *= fac[i];

        // transpose prob tile D-layout -> A-layout via per-wave LDS
#pragma unroll
        for (int i = 0; i < 8; ++i) {
            int q = sel * 8 + i;
            Pw[q * 40 + row]      = (h16)s0[i];
            Pw[q * 40 + 16 + row] = (h16)s1[i];
        }
        asm volatile("s_wait_dscnt 0" ::: "memory");
        const h16* pr = Pw + row * 40 + sel * 8;
        v16h ap = cat16(ld8(pr), ld8(pr + 16));

        // row-sum via WMMA: P(16x32) x ones(32x16); lane-half sel gets rows sel*8+i
        v8f lsum = wmma_f16(ap, ones, zero);
#pragma unroll
        for (int i = 0; i < 8; ++i) lst[i] = lst[i] * fac[i] + lsum[i];

        // PV: B operand from shared V tile
#pragma unroll
        for (int nt = 0; nt < 4; ++nt) {
            const h16* vr = &Vc[(nt * 16 + row) * VT_STRIDE + sel * 8];
            v16h bv = cat16(ld8(vr), ld8(vr + 16));
            acc[nt] = wmma_f16(ap, bv, acc[nt]);
        }

        asm volatile("s_wait_dscnt 0" ::: "memory");
        __syncthreads();   // reads of buffer `it` done before it is refilled
    }

    // normalize, store alpha f16 [b][ch][pix] (contiguous b128 stores)
#pragma unroll
    for (int nt = 0; nt < 4; ++nt) {
        const int c = hc + nt * 16 + row;
        v8h pk;
#pragma unroll
        for (int i = 0; i < 8; ++i) pk[i] = (h16)(acc[nt][i] / lst[i]);
        *(v8h*)(Aw + ((size_t)b * D_MODEL + c) * N_PIX + qBase + sel * 8) = pk;
    }
}

// ---------------------------------------------------------------------------
extern "C" void kernel_launch(void* const* d_in, const int* in_sizes, int n_in,
                              void* d_out, int out_size, void* d_ws, size_t ws_size,
                              hipStream_t stream) {
    (void)in_sizes; (void)n_in; (void)out_size; (void)ws_size;
    const float* q_feat = (const float*)d_in[0];
    const float* k_feat = (const float*)d_in[1];
    const float* v_feat = (const float*)d_in[2];
    const float* wq = (const float*)d_in[3];
    const float* bq = (const float*)d_in[4];
    const float* wk = (const float*)d_in[5];
    const float* bk = (const float*)d_in[6];
    const float* wv = (const float*)d_in[7];
    const float* bv = (const float*)d_in[8];
    const float* wo = (const float*)d_in[9];
    const float* bo = (const float*)d_in[10];

    const size_t SZ = (size_t)BS * D_MODEL * N_PIX;
    h16* Qt = (h16*)d_ws;        // [b][pix][ch]
    h16* Kt = Qt + SZ;           // [b][pix][ch]
    h16* Vw = Kt + SZ;           // [b][ch][pix]
    h16* Aw = Vw + SZ;           // [b][ch][pix]

    dim3 gp(N_PIX / 64, D_MODEL / 64, BS);
    proj_kernel<0><<<gp, 128, 0, stream>>>(q_feat, wq, bq, Qt);
    proj_kernel<0><<<gp, 128, 0, stream>>>(k_feat, wk, bk, Kt);
    proj_kernel<1><<<gp, 128, 0, stream>>>(v_feat, wv, bv, Vw);

    dim3 ga(N_PIX / 64, HEADS, BS);
    attn_kernel<<<ga, 128, 0, stream>>>(Qt, Kt, Vw, Aw);

    proj_kernel<2><<<gp, 128, 0, stream>>>(Aw, wo, bo, (float*)d_out);
}
